// Phi4VisionSiglip2Embeddings_41626823033111
// MI455X (gfx1250) — matrix operand order repaired and checked by reading, not compiled, not run on hardware
//
#include <hip/hip_runtime.h>

// ---------------------------------------------------------------------------
// Problem constants (static shapes from the reference)
// ---------------------------------------------------------------------------
#define BATCH   16
#define MAXLEN  4096
#define KDIM    768      // C*P*P
#define NDIM    1152     // D
#define SGRID   16       // pos-embed grid side

// GEMM tiling: block 128x128, K-step 32, 8 waves (4M x 2N), wave tile 32x64
#define TM  128
#define TN  128
#define TK  32
#define NT  (KDIM / TK)  // 24 K-steps
#define LDA 40           // padded LDS stride (bf16 elems): 80B = 20 banks
#define LDB 40

typedef __attribute__((ext_vector_type(16))) __bf16 v16bf;
typedef __attribute__((ext_vector_type(8)))  float  v8f;

// pack two f32 -> packed bf16x2 (low = a, high = b)
static __device__ __forceinline__ unsigned pack_bf16(float a, float b) {
#if __has_builtin(__builtin_amdgcn_cvt_pk_bf16_f32)
  typedef __attribute__((ext_vector_type(2))) __bf16 v2bf;
  union { v2bf v; unsigned u; } c;
  c.v = __builtin_amdgcn_cvt_pk_bf16_f32(a, b);
  return c.u;
#else
  // native fptrunc f32->bf16 (RNE); selects hardware cvt when available
  union { __bf16 h[2]; unsigned u; } c;
  c.h[0] = (__bf16)a;
  c.h[1] = (__bf16)b;
  return c.u;
#endif
}

// ---------------------------------------------------------------------------
// Kernel 1: per-sample antialiased bilinear resize of pos embeddings -> d_out
//   matches jax.image.resize(method="linear", antialias=True):
//   src = (o+0.5)*(in/out)-0.5 ; w_i = max(0, 1-|i-src|*min(out/in,1)),
//   in-range taps only, row-normalized. Pad region uses resized[0] (y=x=0).
// ---------------------------------------------------------------------------
static __device__ __forceinline__ void taps(int o, int outsz, int* lo_out,
                                            float* w /*[6]*/) {
  float inv     = (float)SGRID / (float)outsz;        // in/out
  float src     = ((float)o + 0.5f) * inv - 0.5f;
  float kscale  = fminf((float)outsz / (float)SGRID, 1.0f);
  float support = 1.0f / kscale;                      // <= 2  -> <= 5 taps
  int   lo      = (int)ceilf(src - support);
  float sum = 0.0f;
#pragma unroll
  for (int i = 0; i < 6; ++i) {
    int   idx = lo + i;
    float t   = 1.0f - fabsf((float)idx - src) * kscale;
    float wv  = (idx >= 0 && idx < SGRID) ? fmaxf(t, 0.0f) : 0.0f;
    w[i] = wv;
    sum += wv;
  }
  float rs = 1.0f / sum;
#pragma unroll
  for (int i = 0; i < 6; ++i) w[i] *= rs;
  *lo_out = lo;
}

__global__ __launch_bounds__(256) void pos_resize_kernel(
    const float* __restrict__ pe,      // [S*S, NDIM]
    const int*   __restrict__ shapes,  // [B, 2]
    float*       __restrict__ out) {   // [B, MAXLEN, NDIM]
  const int b = blockIdx.x >> 12;          // / MAXLEN
  const int l = blockIdx.x & (MAXLEN - 1);
  const int h = shapes[2 * b + 0];
  const int w = shapes[2 * b + 1];

  int y = 0, x = 0;
  if (l < h * w) { y = l / w; x = l - y * w; }

  float wy[6], wx[6];
  int ylo, xlo;
  taps(y, h, &ylo, wy);
  taps(x, w, &xlo, wx);

  float* __restrict__ orow = out + ((size_t)b * MAXLEN + l) * NDIM;
  for (int d = threadIdx.x; d < NDIM; d += 256) {
    float acc = 0.0f;
#pragma unroll
    for (int i = 0; i < 6; ++i) {
      float wyi = wy[i];
      if (wyi == 0.0f) continue;                     // wave-uniform skip
      int yi = min(max(ylo + i, 0), SGRID - 1);
      const float* __restrict__ prow = pe + ((size_t)yi * SGRID) * NDIM + d;
      float racc = 0.0f;
#pragma unroll
      for (int j = 0; j < 6; ++j) {
        float wxj = wx[j];
        if (wxj == 0.0f) continue;
        int xj = min(max(xlo + j, 0), SGRID - 1);
        racc += wxj * prow[(size_t)xj * NDIM];       // pe is L2-hot (1.2 MB)
      }
      acc += wyi * racc;
    }
    orow[d] = acc;
  }
}

// ---------------------------------------------------------------------------
// Kernel 2: bf16 WMMA GEMM  out += A @ W + bias   (out already holds pos)
//   Double-buffered LDS (one barrier/iter), register prefetch of next tile.
//   Fragment layouts (wave32, ISA 7.12.2):
//     A: lane(0-15)=row, K blocks [8h,8h+8) and [8h+16,8h+24), h=lane>>4
//     B (stored N-major in LDS): lane(0-15)=col, K blocks [8h,8h+8) and [8h+16,8h+24)
// ---------------------------------------------------------------------------
__global__ __launch_bounds__(256) void gemm_bias_add_kernel(
    const float* __restrict__ A,     // [M, KDIM]
    const float* __restrict__ W,     // [KDIM, NDIM]
    const float* __restrict__ bias,  // [NDIM]
    float*       __restrict__ out) { // [M, NDIM], RMW (holds pos)
  __shared__ unsigned short As[2][TM * LDA];   // 2 x 10 KB
  __shared__ unsigned short Bs[2][TN * LDB];   // 2 x 10 KB

  const int tid  = threadIdx.x;
  const int m0   = blockIdx.y * TM;
  const int n0   = blockIdx.x * TN;
  const int wave = tid >> 5;
  const int lane = tid & 31;
  const int wm   = wave >> 1;      // 0..3 -> M offset wm*32
  const int wn   = wave & 1;       // 0..1 -> N offset wn*64
  const int r    = lane & 15;
  const int half = lane >> 4;

  // A loader: idx = tid + i*256 (i<4): row = idx>>3, col4 = (idx&7)*4
  const int a_row  = tid >> 3;           // base row for i=0 (rows advance by 32)
  const int a_col  = (tid & 7) << 2;
  // B loader: idx = tid + i*256 (i<2): k2 = idx>>5 (K pair), n = (idx&31)*4
  const int b_k2   = tid >> 5;           // base k-pair for i=0 (advances by 8)
  const int b_n    = (tid & 31) << 2;

  v8f acc[2][4] = {};

  float4 aReg[4];
  float4 bReg0[2], bReg1[2];

  // ---- prologue: load + stage tile 0 ----
#pragma unroll
  for (int i = 0; i < 4; ++i) {
    int row = a_row + i * 32;
    aReg[i] = *(const float4*)(A + (size_t)(m0 + row) * KDIM + a_col);
  }
#pragma unroll
  for (int i = 0; i < 2; ++i) {
    int k = (b_k2 + i * 8) * 2;
    bReg0[i] = *(const float4*)(W + (size_t)(k + 0) * NDIM + (n0 + b_n));
    bReg1[i] = *(const float4*)(W + (size_t)(k + 1) * NDIM + (n0 + b_n));
  }
#pragma unroll
  for (int i = 0; i < 4; ++i) {
    int row = a_row + i * 32;
    uint2 p;
    p.x = pack_bf16(aReg[i].x, aReg[i].y);
    p.y = pack_bf16(aReg[i].z, aReg[i].w);
    *(uint2*)(&As[0][row * LDA + a_col]) = p;
  }
#pragma unroll
  for (int i = 0; i < 2; ++i) {
    int k2 = b_k2 + i * 8;
    unsigned* dst = (unsigned*)&Bs[0][0];
    // column-major (N-major): Bs[n*LDB + 2*k2], packed K pair per store
    dst[((b_n + 0) * LDB >> 1) + k2] = pack_bf16(bReg0[i].x, bReg1[i].x);
    dst[((b_n + 1) * LDB >> 1) + k2] = pack_bf16(bReg0[i].y, bReg1[i].y);
    dst[((b_n + 2) * LDB >> 1) + k2] = pack_bf16(bReg0[i].z, bReg1[i].z);
    dst[((b_n + 3) * LDB >> 1) + k2] = pack_bf16(bReg0[i].w, bReg1[i].w);
  }
  __syncthreads();

  for (int t = 0; t < NT; ++t) {
    const int cur = t & 1;
    const bool more = (t + 1) < NT;
    const int knext = (t + 1) * TK;

    // ---- register-prefetch next tile (overlaps with WMMAs below) ----
    if (more) {
#pragma unroll
      for (int i = 0; i < 4; ++i) {
        int row = a_row + i * 32;
        const float* gp = A + (size_t)(m0 + row) * KDIM + knext + a_col;
        aReg[i] = *(const float4*)gp;
        if (t + 2 < NT) __builtin_prefetch(gp + TK, 0, 1);  // global_prefetch_b8
      }
#pragma unroll
      for (int i = 0; i < 2; ++i) {
        int k = knext + (b_k2 + i * 8) * 2;
        bReg0[i] = *(const float4*)(W + (size_t)(k + 0) * NDIM + (n0 + b_n));
        bReg1[i] = *(const float4*)(W + (size_t)(k + 1) * NDIM + (n0 + b_n));
      }
    }

    // ---- fragments + WMMA from current buffer ----
    v16bf afrag[2], bfrag[4];
#pragma unroll
    for (int mi = 0; mi < 2; ++mi) {
      int row = wm * 32 + mi * 16 + r;
      int kb  = half * 8;
      union { v16bf v; uint4 q[2]; } u;
      u.q[0] = *(const uint4*)(&As[cur][row * LDA + kb]);       // K kb..kb+7
      u.q[1] = *(const uint4*)(&As[cur][row * LDA + kb + 16]);  // K kb+16..+23
      afrag[mi] = u.v;
    }
#pragma unroll
    for (int ni = 0; ni < 4; ++ni) {
      int n = wn * 64 + ni * 16 + r;
      union { v16bf v; uint4 q[2]; } u;
      const unsigned short* p = &Bs[cur][n * LDB + half * 8];  // K 8h..8h+7, 8h+16..8h+23
      u.q[0] = *(const uint4*)(p);
      u.q[1] = *(const uint4*)(p + 16);
      bfrag[ni] = u.v;
    }
#pragma unroll
    for (int mi = 0; mi < 2; ++mi)
#pragma unroll
      for (int ni = 0; ni < 4; ++ni)
        acc[mi][ni] = __builtin_amdgcn_wmma_f32_16x16x32_bf16(
            false, afrag[mi], false, bfrag[ni],
            (short)0, acc[mi][ni], false, false);

    // ---- stage next tile into the other buffer, single barrier ----
    if (more) {
      const int nxt = cur ^ 1;
#pragma unroll
      for (int i = 0; i < 4; ++i) {
        int row = a_row + i * 32;
        uint2 p;
        p.x = pack_bf16(aReg[i].x, aReg[i].y);
        p.y = pack_bf16(aReg[i].z, aReg[i].w);
        *(uint2*)(&As[nxt][row * LDA + a_col]) = p;
      }
#pragma unroll
      for (int i = 0; i < 2; ++i) {
        int k2 = b_k2 + i * 8;
        unsigned* dst = (unsigned*)&Bs[nxt][0];
        dst[((b_n + 0) * LDB >> 1) + k2] = pack_bf16(bReg0[i].x, bReg1[i].x);
        dst[((b_n + 1) * LDB >> 1) + k2] = pack_bf16(bReg0[i].y, bReg1[i].y);
        dst[((b_n + 2) * LDB >> 1) + k2] = pack_bf16(bReg0[i].z, bReg1[i].z);
        dst[((b_n + 3) * LDB >> 1) + k2] = pack_bf16(bReg0[i].w, bReg1[i].w);
      }
    }
    __syncthreads();
  }

  // ---- epilogue: out = acc + bias + out(pos) ----
  // C/D layout: lane(0-15)=col N, VGPR g -> row g (lanes 0-15) / 8+g (16-31)
#pragma unroll
  for (int ni = 0; ni < 4; ++ni) {
    int n    = n0 + wn * 64 + ni * 16 + r;
    float bv = bias[n];
#pragma unroll
    for (int mi = 0; mi < 2; ++mi) {
      int mbase = m0 + wm * 32 + mi * 16 + half * 8;
#pragma unroll
      for (int g = 0; g < 8; ++g) {
        size_t o = (size_t)(mbase + g) * NDIM + n;
        out[o] = out[o] + acc[mi][ni][g] + bv;
      }
    }
  }
}

// ---------------------------------------------------------------------------
// launch
// ---------------------------------------------------------------------------
extern "C" void kernel_launch(void* const* d_in, const int* in_sizes, int n_in,
                              void* d_out, int out_size, void* d_ws, size_t ws_size,
                              hipStream_t stream) {
  const float* pixel_values = (const float*)d_in[0];  // [16,4096,768]
  const int*   spatial      = (const int*)  d_in[1];  // [16,2]
  const float* patch_weight = (const float*)d_in[2];  // [768,1152]
  const float* patch_bias   = (const float*)d_in[3];  // [1152]
  const float* pos_embed    = (const float*)d_in[4];  // [256,1152]
  float*       out          = (float*)d_out;          // [16,4096,1152]
  (void)in_sizes; (void)n_in; (void)out_size; (void)d_ws; (void)ws_size;

  // 1) pos embedding resize -> d_out (fully overwrites; deterministic)
  pos_resize_kernel<<<dim3(BATCH * MAXLEN), dim3(256), 0, stream>>>(
      pos_embed, spatial, out);

  // 2) bf16 WMMA GEMM, epilogue adds bias + pos (RMW of d_out)
  dim3 grid(NDIM / TN, (BATCH * MAXLEN) / TM);  // (9, 512)
  gemm_bias_add_kernel<<<grid, dim3(256), 0, stream>>>(
      pixel_values, patch_weight, patch_bias, out);
}
